// GatedDeltaNet_82858509074962
// MI455X (gfx1250) — compile-verified, matrix operand order checked by reading, not run on hardware
//
#include <hip/hip_runtime.h>
#include <hip/hip_bf16.h>

typedef __attribute__((ext_vector_type(16))) _Float16 v16h;
typedef __attribute__((ext_vector_type(8)))  _Float16 v8h;
typedef __attribute__((ext_vector_type(8)))  float    v8f;
typedef __attribute__((ext_vector_type(4)))  int      v4i;

#ifndef __has_builtin
#define __has_builtin(x) 0
#endif
#if __has_builtin(__builtin_amdgcn_global_load_async_to_lds_b128)
#define HAVE_ASYNC_LDS 1
#else
#define HAVE_ASYNC_LDS 0
#endif
#define AS1 __attribute__((address_space(1)))
#define AS3 __attribute__((address_space(3)))

#if HAVE_ASYNC_LDS
#if __has_builtin(__builtin_amdgcn_s_wait_asynccnt)
#define ASYNC_WAIT_N(n) __builtin_amdgcn_s_wait_asynccnt(n)
#else
#define ASYNC_WAIT_N(n) asm volatile("s_wait_asynccnt %0" ::"i"(n) : "memory")
#endif
// builtin expects (v4i AS1* gsrc, v4i AS3* ldst, imm offset, imm cpol)
#define STAGE_B128(gptr, lptr)                                          \
  __builtin_amdgcn_global_load_async_to_lds_b128(                       \
      (AS1 v4i*)(AS1 void*)(gptr), (AS3 v4i*)(AS3 void*)(lptr), 0, 0)
#else
#define ASYNC_WAIT_N(n) ((void)0)
#define STAGE_B128(gptr, lptr) (*(v8h*)(lptr) = *(const v8h*)(gptr))
#endif

// Problem constants (from reference)
constexpr int Bc = 4, Tc = 2048, Dc = 1024, Hc = 8;
constexpr int DK = 96, DV = 192;
constexpr int KD = Hc * DK;     // 768
constexpr int VD = Hc * DV;     // 1536
constexpr int Ic = 4096;
constexpr int Mrows = Bc * Tc;  // 8192

// GEMM tiling
constexpr int BM = 64, BN = 128, BK = 32;
constexpr int LDA = 40;  // padded LDS stride (halves)
constexpr int LDB = 40;

__device__ __forceinline__ float fast_rcp(float x) { return __builtin_amdgcn_rcpf(x); }
__device__ __forceinline__ float fast_sigmoid(float x) { return fast_rcp(1.f + __expf(-x)); }
__device__ __forceinline__ float fast_silu(float x) { return x * fast_sigmoid(x); }

// -------------------------------- f32[K,N] -> f16[N,K] (transpose + convert)
__global__ __launch_bounds__(256)
void k_cvt16t(const float* __restrict__ s, _Float16* __restrict__ d, int N, int K) {
  __shared__ float tile[32][33];
  const int n0 = blockIdx.x * 32, k0 = blockIdx.y * 32;
  const int tx = threadIdx.x & 31, ty = threadIdx.x >> 5;  // 32 x 8
#pragma unroll
  for (int i = 0; i < 32; i += 8)
    tile[ty + i][tx] = s[(size_t)(k0 + ty + i) * N + n0 + tx];
  __syncthreads();
#pragma unroll
  for (int i = 0; i < 32; i += 8)
    d[(size_t)(n0 + ty + i) * K + k0 + tx] = (_Float16)tile[tx][ty + i];
}

// ----------------------------------------------------- RMSNorm row -> f16 out
__global__ __launch_bounds__(256)
void k_rmsnorm16(const float* __restrict__ x, const float* __restrict__ w,
                 _Float16* __restrict__ out) {
  __shared__ float red[8];
  const int row = blockIdx.x;
  const float* px = x + (size_t)row * Dc;
  float ss = 0.f;
#pragma unroll
  for (int i = 0; i < 4; ++i) {
    float v = px[threadIdx.x + i * 256];
    ss += v * v;
  }
  for (int off = 16; off; off >>= 1) ss += __shfl_down(ss, off, 32);
  if ((threadIdx.x & 31) == 0) red[threadIdx.x >> 5] = ss;
  __syncthreads();
  if (threadIdx.x == 0) {
    float tot = 0.f;
#pragma unroll
    for (int i = 0; i < 8; ++i) tot += red[i];
    red[0] = rsqrtf(tot / (float)Dc + 1e-6f);
  }
  __syncthreads();
  const float r = red[0];
#pragma unroll
  for (int i = 0; i < 4; ++i) {
    int c = threadIdx.x + i * 256;
    out[(size_t)row * Dc + c] = (_Float16)(px[c] * r * w[c]);
  }
}

// ------------------------------------------------------------ WMMA GEMM core
// C[M,N] (f32) = A[M,K](f16,row-major) * Bt[N,K](f16, pre-transposed) [+ resid]
// Block: 256 thr = 8 waves (2 M x 4 N), wave tile 32x32, 2x2 WMMA accum.
// Double-buffered LDS tiles; async global->LDS staged one K-tile ahead.
template <bool RESID>
__global__ __launch_bounds__(256)
void k_gemm(const _Float16* __restrict__ A, const _Float16* __restrict__ Bt,
            float* __restrict__ C, const float* __restrict__ R,
            int Ndim, int Kdim) {
  __shared__ _Float16 As[2][BM * LDA];
  __shared__ _Float16 Bs[2][BN * LDB];
  const int tid = threadIdx.x;
  const int lane = tid & 31, wave = tid >> 5;
  const int wm = wave & 1, wn = wave >> 1;
  const int m0 = blockIdx.y * BM, n0 = blockIdx.x * BN;

  v8f acc[2][2] = {};
  const int arow = tid >> 2;
  const int acol = (tid & 3) * 8;
  const int hrow = lane & 15, kg = lane >> 4;

  auto stage = [&](int buf, int kk) {  // 3 async b128 loads per thread
    STAGE_B128(A + (size_t)(m0 + arow) * Kdim + kk + acol, &As[buf][arow * LDA + acol]);
#pragma unroll
    for (int j = 0; j < 2; ++j) {
      int c = tid + j * 256;
      int nn = c >> 2, kc = (c & 3) * 8;
      STAGE_B128(Bt + (size_t)(n0 + nn) * Kdim + kk + kc, &Bs[buf][nn * LDB + kc]);
    }
  };

  stage(0, 0);
  int cur = 0;
  for (int k0 = 0; k0 < Kdim; k0 += BK) {
    const bool has_next = (k0 + BK) < Kdim;
    if (has_next) stage(cur ^ 1, k0 + BK);          // tile i+1 in flight
    if (has_next) ASYNC_WAIT_N(3); else ASYNC_WAIT_N(0);  // tile i landed
    __syncthreads();

    v16h af[2], bf[2];
#pragma unroll
    for (int mr = 0; mr < 2; ++mr) {
      const _Float16* p = &As[cur][(wm * 32 + mr * 16 + hrow) * LDA];
      v8h lo = *(const v8h*)(p + kg * 8);
      v8h hi = *(const v8h*)(p + 16 + kg * 8);
#pragma unroll
      for (int e = 0; e < 8; ++e) { af[mr][e] = lo[e]; af[mr][e + 8] = hi[e]; }
    }
#pragma unroll
    for (int nc = 0; nc < 2; ++nc) {
      const _Float16* p = &Bs[cur][(wn * 32 + nc * 16 + hrow) * LDB];
      v8h lo = *(const v8h*)(p + kg * 8);
      v8h hi = *(const v8h*)(p + 16 + kg * 8);
#pragma unroll
      for (int e = 0; e < 8; ++e) { bf[nc][e] = lo[e]; bf[nc][e + 8] = hi[e]; }
    }
#pragma unroll
    for (int mr = 0; mr < 2; ++mr)
#pragma unroll
      for (int nc = 0; nc < 2; ++nc)
        acc[mr][nc] = __builtin_amdgcn_wmma_f32_16x16x32_f16(
            false, af[mr], false, bf[nc], (short)0, acc[mr][nc], false, false);
    if (has_next) __syncthreads();  // reads done before tile i+2 overwrites
    cur ^= 1;
  }

  const int half = lane >> 4;
#pragma unroll
  for (int mr = 0; mr < 2; ++mr)
#pragma unroll
    for (int nc = 0; nc < 2; ++nc) {
      int col = n0 + wn * 32 + nc * 16 + hrow;
#pragma unroll
      for (int r = 0; r < 8; ++r) {
        int row = m0 + wm * 32 + mr * 16 + half * 8 + r;
        size_t idx = (size_t)row * Ndim + col;
        float v = acc[mr][nc][r];
        if (RESID) v += R[idx];
        C[idx] = v;
      }
    }
}

// ---------------------------------------- fused SwiGLU GEMM: silu(A@Bg)*(A@Bu)
__global__ __launch_bounds__(256)
void k_swiglu_gemm(const _Float16* __restrict__ A, const _Float16* __restrict__ Bgt,
                   const _Float16* __restrict__ But, _Float16* __restrict__ out,
                   int Ndim, int Kdim) {
  __shared__ _Float16 As[2][BM * LDA];
  __shared__ _Float16 Bsg[2][BN * LDB];
  __shared__ _Float16 Bsu[2][BN * LDB];
  const int tid = threadIdx.x;
  const int lane = tid & 31, wave = tid >> 5;
  const int wm = wave & 1, wn = wave >> 1;
  const int m0 = blockIdx.y * BM, n0 = blockIdx.x * BN;

  v8f accg[2][2] = {}, accu[2][2] = {};
  const int arow = tid >> 2;
  const int acol = (tid & 3) * 8;
  const int hrow = lane & 15, kg = lane >> 4;

  auto stage = [&](int buf, int kk) {  // 5 async b128 loads per thread
    STAGE_B128(A + (size_t)(m0 + arow) * Kdim + kk + acol, &As[buf][arow * LDA + acol]);
#pragma unroll
    for (int j = 0; j < 2; ++j) {
      int c = tid + j * 256;
      int nn = c >> 2, kc = (c & 3) * 8;
      STAGE_B128(Bgt + (size_t)(n0 + nn) * Kdim + kk + kc, &Bsg[buf][nn * LDB + kc]);
      STAGE_B128(But + (size_t)(n0 + nn) * Kdim + kk + kc, &Bsu[buf][nn * LDB + kc]);
    }
  };

  stage(0, 0);
  int cur = 0;
  for (int k0 = 0; k0 < Kdim; k0 += BK) {
    const bool has_next = (k0 + BK) < Kdim;
    if (has_next) stage(cur ^ 1, k0 + BK);
    if (has_next) ASYNC_WAIT_N(5); else ASYNC_WAIT_N(0);
    __syncthreads();

    v16h af[2], bfg[2], bfu[2];
#pragma unroll
    for (int mr = 0; mr < 2; ++mr) {
      const _Float16* p = &As[cur][(wm * 32 + mr * 16 + hrow) * LDA];
      v8h lo = *(const v8h*)(p + kg * 8);
      v8h hi = *(const v8h*)(p + 16 + kg * 8);
#pragma unroll
      for (int e = 0; e < 8; ++e) { af[mr][e] = lo[e]; af[mr][e + 8] = hi[e]; }
    }
#pragma unroll
    for (int nc = 0; nc < 2; ++nc) {
      const _Float16* pg = &Bsg[cur][(wn * 32 + nc * 16 + hrow) * LDB];
      const _Float16* pu = &Bsu[cur][(wn * 32 + nc * 16 + hrow) * LDB];
      v8h lg = *(const v8h*)(pg + kg * 8), hg = *(const v8h*)(pg + 16 + kg * 8);
      v8h lu = *(const v8h*)(pu + kg * 8), hu = *(const v8h*)(pu + 16 + kg * 8);
#pragma unroll
      for (int e = 0; e < 8; ++e) {
        bfg[nc][e] = lg[e]; bfg[nc][e + 8] = hg[e];
        bfu[nc][e] = lu[e]; bfu[nc][e + 8] = hu[e];
      }
    }
#pragma unroll
    for (int mr = 0; mr < 2; ++mr)
#pragma unroll
      for (int nc = 0; nc < 2; ++nc) {
        accg[mr][nc] = __builtin_amdgcn_wmma_f32_16x16x32_f16(
            false, af[mr], false, bfg[nc], (short)0, accg[mr][nc], false, false);
        accu[mr][nc] = __builtin_amdgcn_wmma_f32_16x16x32_f16(
            false, af[mr], false, bfu[nc], (short)0, accu[mr][nc], false, false);
      }
    if (has_next) __syncthreads();
    cur ^= 1;
  }

  const int half = lane >> 4;
#pragma unroll
  for (int mr = 0; mr < 2; ++mr)
#pragma unroll
    for (int nc = 0; nc < 2; ++nc) {
      int col = n0 + wn * 32 + nc * 16 + hrow;
#pragma unroll
      for (int r = 0; r < 8; ++r) {
        int row = m0 + wm * 32 + mr * 16 + half * 8 + r;
        float g = accg[mr][nc][r];
        float u = accu[mr][nc][r];
        out[(size_t)row * Ndim + col] = (_Float16)(fast_silu(g) * u);
      }
    }
}

// --------------------------------------------- depthwise causal conv4 + SiLU
__global__ void k_conv_silu(const float* __restrict__ lin, const float* __restrict__ w,
                            float* __restrict__ out, int C, int total) {
  int idx = blockIdx.x * blockDim.x + threadIdx.x;
  if (idx >= total) return;
  int c = idx % C;
  int bt = idx / C;
  int t = bt % Tc;
  float acc = 0.f;
#pragma unroll
  for (int i = 0; i < 4; ++i) {
    int tt = t - 3 + i;
    if (tt >= 0) acc += lin[(size_t)(bt - (3 - i)) * C + c] * w[c * 4 + i];
  }
  out[idx] = fast_silu(acc);
}

// ----------------------------------------------- per-head L2 norm (in place)
__global__ void k_l2norm(float* __restrict__ x, float scale, int ngroups) {
  int g = blockIdx.x * blockDim.x + threadIdx.x;
  if (g >= ngroups) return;
  float* p = x + (size_t)g * DK;
  float ss = 1e-6f;
  for (int i = 0; i < DK; ++i) ss += p[i] * p[i];
  float r = rsqrtf(ss) * scale;
  for (int i = 0; i < DK; ++i) p[i] *= r;
}

// ------------------------------------------- beta / decay projections (N=8)
__global__ __launch_bounds__(256)
void k_beta_g(const _Float16* __restrict__ h16, const float* __restrict__ Wb,
              const float* __restrict__ Wa, const float* __restrict__ A_log,
              const float* __restrict__ dt_bias, float* __restrict__ beta,
              float* __restrict__ gdec) {
  const int row = blockIdx.x;
  const int lane = threadIdx.x & 31, head = threadIdx.x >> 5;  // 8 waves = 8 heads
  const _Float16* ph = h16 + (size_t)row * Dc;
  float sb = 0.f, sa = 0.f;
  for (int i = lane; i < Dc; i += 32) {
    float hv = (float)ph[i];
    sb += hv * Wb[(size_t)i * Hc + head];
    sa += hv * Wa[(size_t)i * Hc + head];
  }
  for (int off = 16; off; off >>= 1) {
    sb += __shfl_down(sb, off, 32);
    sa += __shfl_down(sa, off, 32);
  }
  if (lane == 0) {
    beta[(size_t)row * Hc + head] = fast_sigmoid(sb);
    float xs = sa + dt_bias[head];
    float sp = (xs > 20.f) ? xs : log1pf(__expf(xs));
    gdec[(size_t)row * Hc + head] = -__expf(A_log[head]) * sp;
  }
}

// --------------------------- gated delta-rule scan: one (b,h) per workgroup,
// one state column S[:,j] (96 floats) per thread, fully in registers.
__global__ __launch_bounds__(192)
void k_scan(const float* __restrict__ q, const float* __restrict__ k,
            const float* __restrict__ v, const float* __restrict__ gdec,
            const float* __restrict__ beta, float* __restrict__ o) {
  const int bh = blockIdx.x;
  const int b = bh / Hc, h = bh % Hc;
  const int j = threadIdx.x;  // 0..191 -> dv column
  __shared__ float kb[DK], qb[DK];
  float s[DK];
#pragma unroll
  for (int i = 0; i < DK; ++i) s[i] = 0.f;

  for (int t = 0; t < Tc; ++t) {
    const size_t bt = (size_t)b * Tc + t;
    if (j < DK) {
      size_t base = bt * KD + (size_t)h * DK + j;
      kb[j] = k[base];
      qb[j] = q[base];
    }
    __syncthreads();
    const size_t gi = bt * Hc + h;
    const float ge = __expf(gdec[gi]);
    const float bt_ = beta[gi];
    const float vt = v[bt * VD + (size_t)h * DV + j];
    float vold = 0.f;
#pragma unroll
    for (int i = 0; i < DK; ++i) vold += s[i] * kb[i];
    const float delta = (vt - ge * vold) * bt_;
    float acc = 0.f;
#pragma unroll
    for (int i = 0; i < DK; ++i) {
      s[i] = s[i] * ge + kb[i] * delta;
      acc += qb[i] * s[i];
    }
    o[(bt * Hc + h) * DV + j] = acc;
    __syncthreads();
  }
}

// ------------------------------------- gated RMSNorm of o, silu gate -> f16
__global__ void k_gated_norm(const float* __restrict__ o, const float* __restrict__ glin,
                             const float* __restrict__ onw, _Float16* __restrict__ og16,
                             int ngroups) {
  int g = blockIdx.x * blockDim.x + threadIdx.x;
  if (g >= ngroups) return;
  const float* po = o + (size_t)g * DV;
  const float* pg = glin + (size_t)g * DV;  // VD == Hc*DV, same flat index
  float ss = 0.f;
  for (int i = 0; i < DV; ++i) ss += po[i] * po[i];
  float r = rsqrtf(ss / (float)DV + 1e-6f);
  _Float16* pd = og16 + (size_t)g * DV;
  for (int i = 0; i < DV; ++i) {
    pd[i] = (_Float16)(po[i] * r * onw[i] * fast_silu(pg[i]));
  }
}

// ============================================================== host launch
extern "C" void kernel_launch(void* const* d_in, const int* in_sizes, int n_in,
                              void* d_out, int out_size, void* d_ws, size_t ws_size,
                              hipStream_t stream) {
  const float* x        = (const float*)d_in[0];
  const float* Wq       = (const float*)d_in[1];
  const float* Wk       = (const float*)d_in[2];
  const float* Wv       = (const float*)d_in[3];
  const float* Wb       = (const float*)d_in[4];
  const float* Wa       = (const float*)d_in[5];
  const float* A_log    = (const float*)d_in[6];
  const float* dt_bias  = (const float*)d_in[7];
  const float* conv_q_w = (const float*)d_in[8];
  const float* conv_k_w = (const float*)d_in[9];
  const float* conv_v_w = (const float*)d_in[10];
  const float* Wg       = (const float*)d_in[11];
  const float* onorm_w  = (const float*)d_in[12];
  const float* Wo       = (const float*)d_in[13];
  const float* norm1_w  = (const float*)d_in[14];
  const float* norm2_w  = (const float*)d_in[15];
  const float* Wgate    = (const float*)d_in[16];
  const float* Wup      = (const float*)d_in[17];
  const float* Wdown    = (const float*)d_in[18];
  float* out = (float*)d_out;

  char* ws = (char*)d_ws;
  size_t off = 0;
  auto alloc = [&](size_t bytes) -> void* {
    void* p = ws + off;
    off = (off + bytes + 255) & ~(size_t)255;
    return p;
  };
  _Float16* h16     = (_Float16*)alloc((size_t)Mrows * Dc * 2);
  _Float16* h2_16   = (_Float16*)alloc((size_t)Mrows * Dc * 2);
  _Float16* wq16    = (_Float16*)alloc((size_t)Dc * KD * 2);   // [KD][Dc]
  _Float16* wk16    = (_Float16*)alloc((size_t)Dc * KD * 2);
  _Float16* wv16    = (_Float16*)alloc((size_t)Dc * VD * 2);
  _Float16* wg16    = (_Float16*)alloc((size_t)Dc * VD * 2);
  _Float16* wo16    = (_Float16*)alloc((size_t)VD * Dc * 2);   // [Dc][VD]
  _Float16* wgate16 = (_Float16*)alloc((size_t)Dc * Ic * 2);   // [Ic][Dc]
  _Float16* wup16   = (_Float16*)alloc((size_t)Dc * Ic * 2);
  _Float16* wdown16 = (_Float16*)alloc((size_t)Ic * Dc * 2);   // [Dc][Ic]
  float* qlin = (float*)alloc((size_t)Mrows * KD * 4);
  float* klin = (float*)alloc((size_t)Mrows * KD * 4);
  float* vlin = (float*)alloc((size_t)Mrows * VD * 4);
  float* glin = (float*)alloc((size_t)Mrows * VD * 4);
  float* qbuf = (float*)alloc((size_t)Mrows * KD * 4);
  float* kbuf = (float*)alloc((size_t)Mrows * KD * 4);
  float* vbuf = (float*)alloc((size_t)Mrows * VD * 4);
  float* beta = (float*)alloc((size_t)Mrows * Hc * 4);
  float* gdec = (float*)alloc((size_t)Mrows * Hc * 4);
  float* obuf = (float*)alloc((size_t)Mrows * VD * 4);
  _Float16* og16 = (_Float16*)alloc((size_t)Mrows * VD * 2);
  float* xres = (float*)alloc((size_t)Mrows * Dc * 4);
  _Float16* mlp16 = (_Float16*)alloc((size_t)Mrows * Ic * 2);
  (void)ws_size; (void)n_in; (void)in_sizes; (void)out_size;

  // transpose+convert weights: src [K,N] f32 -> dst [N,K] f16
  auto cvtT = [&](const float* s, _Float16* d, int N, int K) {
    k_cvt16t<<<dim3(N / 32, K / 32), dim3(256), 0, stream>>>(s, d, N, K);
  };
  cvtT(Wq, wq16, KD, Dc);
  cvtT(Wk, wk16, KD, Dc);
  cvtT(Wv, wv16, VD, Dc);
  cvtT(Wg, wg16, VD, Dc);
  cvtT(Wo, wo16, Dc, VD);
  cvtT(Wgate, wgate16, Ic, Dc);
  cvtT(Wup, wup16, Ic, Dc);
  cvtT(Wdown, wdown16, Dc, Ic);

  // --- h = rmsnorm(x) (f16) ---
  k_rmsnorm16<<<dim3(Mrows), dim3(256), 0, stream>>>(x, norm1_w, h16);

  // --- projections via WMMA GEMM ---
  auto gemm = [&](const _Float16* A, const _Float16* Bt, float* C, int N, int K) {
    k_gemm<false><<<dim3(N / BN, Mrows / BM), dim3(256), 0, stream>>>(A, Bt, C, nullptr, N, K);
  };
  gemm(h16, wq16, qlin, KD, Dc);
  gemm(h16, wk16, klin, KD, Dc);
  gemm(h16, wv16, vlin, VD, Dc);
  gemm(h16, wg16, glin, VD, Dc);

  // --- conv + silu ---
  {
    int tq = Mrows * KD, tv = Mrows * VD;
    k_conv_silu<<<dim3((tq + 255) / 256), dim3(256), 0, stream>>>(qlin, conv_q_w, qbuf, KD, tq);
    k_conv_silu<<<dim3((tq + 255) / 256), dim3(256), 0, stream>>>(klin, conv_k_w, kbuf, KD, tq);
    k_conv_silu<<<dim3((tv + 255) / 256), dim3(256), 0, stream>>>(vlin, conv_v_w, vbuf, VD, tv);
  }
  // --- per-head l2norm (q scaled by dk^-0.5) ---
  {
    int ng = Mrows * Hc;
    k_l2norm<<<dim3((ng + 255) / 256), dim3(256), 0, stream>>>(qbuf, rsqrtf((float)DK), ng);
    k_l2norm<<<dim3((ng + 255) / 256), dim3(256), 0, stream>>>(kbuf, 1.0f, ng);
  }
  // --- beta / decay ---
  k_beta_g<<<dim3(Mrows), dim3(256), 0, stream>>>(h16, Wb, Wa, A_log, dt_bias, beta, gdec);

  // --- gated delta-rule scan ---
  k_scan<<<dim3(Bc * Hc), dim3(192), 0, stream>>>(qbuf, kbuf, vbuf, gdec, beta, obuf);

  // --- gated RMSNorm + silu gate -> f16 ---
  {
    int ng = Mrows * Hc;
    k_gated_norm<<<dim3((ng + 255) / 256), dim3(256), 0, stream>>>(obuf, glin, onorm_w, og16, ng);
  }
  // --- xres = x + og @ Wo ---
  k_gemm<true><<<dim3(Dc / BN, Mrows / BM), dim3(256), 0, stream>>>(og16, wo16, xres, x, Dc, VD);

  // --- MLP: h2 = rmsnorm(xres); mlp16 = silu(h2@Wgate)*(h2@Wup); out = xres + mlp16@Wdown
  k_rmsnorm16<<<dim3(Mrows), dim3(256), 0, stream>>>(xres, norm2_w, h2_16);
  k_swiglu_gemm<<<dim3(Ic / BN, Mrows / BM), dim3(256), 0, stream>>>(h2_16, wgate16, wup16, mlp16, Ic, Dc);
  k_gemm<true><<<dim3(Dc / BN, Mrows / BM), dim3(256), 0, stream>>>(mlp16, wdown16, out, xres, Dc, Ic);
}
